// MultiHeadAttention_73443940762162
// MI455X (gfx1250) — matrix operand order repaired and checked by reading, not compiled, not run on hardware
//
#include <hip/hip_runtime.h>
#include <hip/hip_bf16.h>

typedef _Float16 f16;
typedef __attribute__((ext_vector_type(16))) _Float16 v16h;
typedef __attribute__((ext_vector_type(8)))  _Float16 v8h;
typedef __attribute__((ext_vector_type(4)))  _Float16 v4h;
typedef __attribute__((ext_vector_type(8)))  float    v8f;
typedef __attribute__((ext_vector_type(4)))  float    v4f;

#define WMMA16(A, B, C) \
    __builtin_amdgcn_wmma_f32_16x16x32_f16(false, (A), false, (B), (short)0, (C), false, false)

static constexpr int Bsz = 2, S = 2048, D = 1024, H = 16, DK = 64;

// ---- fragment helpers -------------------------------------------------------

static __device__ __forceinline__ v16h mk16(v8h lo, v8h hi) {
    v16h r;
#pragma unroll
    for (int i = 0; i < 8; ++i) { r[i] = lo[i]; r[i + 8] = hi[i]; }
    return r;
}

// A-fragment (16x32 f16): lanes 0-15 hold K {0..15}, lanes 16-31 hold K {16..31}
static __device__ __forceinline__ v16h ldA16(const f16* __restrict__ rowk, int lane) {
    const int o = (lane & 16) ? 16 : 0;
    return mk16(*(const v8h*)(rowk + o), *(const v8h*)(rowk + o + 8));
}

// ---- kernel 0: bulk fp32 -> f16 conversion (bandwidth-bound, runs once) -----
__global__ __launch_bounds__(256) void cvt_f32_f16_kernel(
    const float* __restrict__ src, f16* __restrict__ dst, int n4)
{
    const int i = blockIdx.x * 256 + threadIdx.x;
    if (i < n4) {
        v4f x = ((const v4f*)src)[i];
        v4h y;
#pragma unroll
        for (int j = 0; j < 4; ++j) y[j] = (f16)x[j];
        ((v4h*)dst)[i] = y;
    }
}

// ---- kernel 1: fused Q/K/V projections (X @ W^T + b), all-f16 fragments -----
// one wave computes a 16(seq) x 64(one head) tile of one projection.
// waves: 3 proj * 256 mtiles * 16 heads = 12288 -> 1536 blocks * 8 waves
__global__ __launch_bounds__(256) void qkv_proj_kernel(
    const f16* __restrict__ Xq, const f16* __restrict__ Xk, const f16* __restrict__ Xv,
    const f16* __restrict__ Wq, const float* __restrict__ bq,
    const f16* __restrict__ Wk, const float* __restrict__ bk,
    const f16* __restrict__ Wv, const float* __restrict__ bv,
    f16* __restrict__ q_ws, f16* __restrict__ k_ws, f16* __restrict__ vT_ws)
{
    const int lane = threadIdx.x & 31;
    const int wave = threadIdx.x >> 5;
    const int g    = blockIdx.x * 8 + wave;        // 0..12287
    const int proj = g / 4096;
    const int rem  = g - proj * 4096;
    const int mtile = rem >> 4;                    // 0..255
    const int h     = rem & 15;                    // 0..15

    const f16 *X, *W;
    const float *bias;
    if (proj == 0)      { X = Xq; W = Wq; bias = bq; }
    else if (proj == 1) { X = Xk; W = Wk; bias = bk; }
    else                { X = Xv; W = Wv; bias = bv; }

    const int r  = mtile * 16 + (lane & 15);       // global row (b*S + s)
    const int b  = (mtile * 16) / S;
    const int s0 = mtile * 16 - b * S;
    const f16* xrow = X + (size_t)r * D;
    const int oB = (lane & 16) ? 16 : 0;

    const f16* w0 = W + (size_t)(h * DK + 0 * 16 + (lane & 15)) * D + oB;
    const f16* w1 = W + (size_t)(h * DK + 1 * 16 + (lane & 15)) * D + oB;
    const f16* w2 = W + (size_t)(h * DK + 2 * 16 + (lane & 15)) * D + oB;
    const f16* w3 = W + (size_t)(h * DK + 3 * 16 + (lane & 15)) * D + oB;

    v8f acc0 = {}, acc1 = {}, acc2 = {}, acc3 = {};
    for (int kk0 = 0; kk0 < D; kk0 += 32) {
        const v16h a = ldA16(xrow + kk0, lane);
        acc0 = WMMA16(a, *(const v16h*)(w0 + kk0), acc0);
        acc1 = WMMA16(a, *(const v16h*)(w1 + kk0), acc1);
        acc2 = WMMA16(a, *(const v16h*)(w2 + kk0), acc2);
        acc3 = WMMA16(a, *(const v16h*)(w3 + kk0), acc3);
    }

    v8f accs[4] = {acc0, acc1, acc2, acc3};
    if (proj < 2) {
        f16* out = (proj == 0) ? q_ws : k_ws;   // [B,H,S,DK]
        const size_t rowbase = ((size_t)(b * H + h) * S + s0);
#pragma unroll
        for (int t = 0; t < 4; ++t) {
            const float bv_ = bias[h * DK + t * 16 + (lane & 15)];
#pragma unroll
            for (int j = 0; j < 8; ++j) {
                const int M = (lane & 16) ? j + 8 : j;
                out[(rowbase + M) * DK + t * 16 + (lane & 15)] = (f16)(accs[t][j] + bv_);
            }
        }
    } else {
        // V stored transposed: [B,H,DK,S] so PV B-fragments are contiguous
#pragma unroll
        for (int t = 0; t < 4; ++t) {
            const float bv_ = bias[h * DK + t * 16 + (lane & 15)];
            v8h pk;
#pragma unroll
            for (int j = 0; j < 8; ++j) pk[j] = (f16)(accs[t][j] + bv_);
            f16* dst = vT_ws + ((size_t)(b * H + h) * DK + t * 16 + (lane & 15)) * S
                             + s0 + ((lane & 16) ? 8 : 0);
            *(v8h*)dst = pk;
        }
    }
}

// ---- kernel 2: flash attention ----------------------------------------------
// one wave owns a 16-row q tile of one (b,h); 8 waves/block.
// waves: B*H*(S/16) = 4096 -> 512 blocks
__global__ __launch_bounds__(256) void attn_kernel(
    const f16* __restrict__ q_ws, const f16* __restrict__ k_ws,
    const f16* __restrict__ vT_ws, const int* __restrict__ padding_mask,
    f16* __restrict__ att_ws)
{
    __shared__ _Float16 lds_p[8][16 * 32];   // per-wave P tile (D-layout -> A-layout bounce)

    const int lane = threadIdx.x & 31;
    const int wave = threadIdx.x >> 5;
    const int g    = blockIdx.x * 8 + wave;  // 0..4095
    const int bh   = g >> 7;                 // 0..31
    const int qt   = g & 127;
    const int b    = bh >> 4;
    const int h    = bh & 15;
    const int q0   = qt * 16;
    const int row  = lane & 15;
    const int oA   = (lane & 16) ? 16 : 0;
    const int oB   = (lane & 16) ? 16 : 0;

    const f16* qbase = q_ws  + (size_t)bh * S * DK;
    const f16* kbase = k_ws  + (size_t)bh * S * DK;
    const f16* vbase = vT_ws + (size_t)bh * DK * S;

    // load q A-fragments once (K = 0..31 and 32..63)
    const f16* qrow = qbase + (size_t)(q0 + row) * DK;
    const v16h qa0 = mk16(*(const v8h*)(qrow + oA),      *(const v8h*)(qrow + oA + 8));
    const v16h qa1 = mk16(*(const v8h*)(qrow + 32 + oA), *(const v8h*)(qrow + 32 + oA + 8));

    v8f m_run, l_run, acc0 = {}, acc1 = {}, acc2 = {}, acc3 = {};
#pragma unroll
    for (int j = 0; j < 8; ++j) { m_run[j] = -3.0e38f; l_run[j] = 0.0f; }

    for (int j0 = 0; j0 <= q0 + 15; j0 += 32) {
        // ---- scores: 16q x 32k in two 16x16 D-fragments, K=DK=64 in 2 WMMAs each
        const f16* kr0 = kbase + (size_t)(j0 + row) * DK;
        const f16* kr1 = kbase + (size_t)(j0 + 16 + row) * DK;
        const v8f z = {};
        v8f s0 = WMMA16(qa0, *(const v16h*)(kr0 + oB), z);
        s0     = WMMA16(qa1, *(const v16h*)(kr0 + 32 + oB), s0);
        v8f s1 = WMMA16(qa0, *(const v16h*)(kr1 + oB), z);
        s1     = WMMA16(qa1, *(const v16h*)(kr1 + 32 + oB), s1);

        const int key0 = j0 + row;
        const int key1 = key0 + 16;
        const int pm0  = padding_mask[b * S + key0];
        const int pm1  = padding_mask[b * S + key1];

        // ---- scale + mask + online softmax (row stats via 16-lane xor reductions)
#pragma unroll
        for (int j = 0; j < 8; ++j) {
            const int qg = q0 + ((lane & 16) ? j + 8 : j);
            float x0 = s0[j] * 0.125f;   // 1/sqrt(64)
            float x1 = s1[j] * 0.125f;
            if (key0 > qg || pm0 == 0) x0 = -1.0e30f;
            if (key1 > qg || pm1 == 0) x1 = -1.0e30f;

            float t = fmaxf(x0, x1);
            t = fmaxf(t, __shfl_xor(t, 1, 32));
            t = fmaxf(t, __shfl_xor(t, 2, 32));
            t = fmaxf(t, __shfl_xor(t, 4, 32));
            t = fmaxf(t, __shfl_xor(t, 8, 32));

            const float mn    = fmaxf(m_run[j], t);
            const float alpha = __expf(m_run[j] - mn);
            m_run[j] = mn;

            const float p0 = __expf(x0 - mn);
            const float p1 = __expf(x1 - mn);
            float rs = p0 + p1;
            rs += __shfl_xor(rs, 1, 32);
            rs += __shfl_xor(rs, 2, 32);
            rs += __shfl_xor(rs, 4, 32);
            rs += __shfl_xor(rs, 8, 32);
            l_run[j] = l_run[j] * alpha + rs;

            acc0[j] *= alpha; acc1[j] *= alpha; acc2[j] *= alpha; acc3[j] *= alpha;

            // stash P (f16) in D-layout: [row M][key]
            const int M = (lane & 16) ? j + 8 : j;
            lds_p[wave][M * 32 + row]      = (f16)p0;
            lds_p[wave][M * 32 + 16 + row] = (f16)p1;
        }

        asm volatile("s_wait_dscnt 0" ::: "memory");

        // ---- reload P as 16x32 A-fragment
        const _Float16* lp = &lds_p[wave][0];
        const v16h pa = mk16(*(const v8h*)(lp + row * 32 + oA),
                             *(const v8h*)(lp + row * 32 + oA + 8));

        // ---- O += P @ V  (B-fragments contiguous thanks to vT layout)
        const f16* vr0 = vbase + (size_t)(0 * 16 + row) * S + j0 + oB;
        const f16* vr1 = vbase + (size_t)(1 * 16 + row) * S + j0 + oB;
        const f16* vr2 = vbase + (size_t)(2 * 16 + row) * S + j0 + oB;
        const f16* vr3 = vbase + (size_t)(3 * 16 + row) * S + j0 + oB;
        acc0 = WMMA16(pa, *(const v16h*)vr0, acc0);
        acc1 = WMMA16(pa, *(const v16h*)vr1, acc1);
        acc2 = WMMA16(pa, *(const v16h*)vr2, acc2);
        acc3 = WMMA16(pa, *(const v16h*)vr3, acc3);
    }

    // ---- normalize and write att [B,S,D] as f16 for the output projection
    v8f accs[4] = {acc0, acc1, acc2, acc3};
#pragma unroll
    for (int t = 0; t < 4; ++t) {
#pragma unroll
        for (int j = 0; j < 8; ++j) {
            const int M = (lane & 16) ? j + 8 : j;
            const float inv = 1.0f / l_run[j];
            att_ws[(size_t)(b * S + q0 + M) * D + h * DK + t * 16 + row] =
                (f16)(accs[t][j] * inv);
        }
    }
}

// ---- kernel 3: output projection (att @ Wo^T + bo) -> f32 d_out -------------
// one wave computes a 16 x 64 output tile. waves: 256*16 = 4096 -> 512 blocks
__global__ __launch_bounds__(256) void out_proj_kernel(
    const f16* __restrict__ att_ws, const f16* __restrict__ Wo,
    const float* __restrict__ bo, float* __restrict__ out)
{
    const int lane  = threadIdx.x & 31;
    const int wave  = threadIdx.x >> 5;
    const int g     = blockIdx.x * 8 + wave;   // 0..4095
    const int mtile = g >> 4;
    const int tg    = g & 15;
    const int oB    = (lane & 16) ? 16 : 0;

    const f16* arow = att_ws + (size_t)(mtile * 16 + (lane & 15)) * D;
    const f16* w0 = Wo + (size_t)(tg * 64 + 0 * 16 + (lane & 15)) * D + oB;
    const f16* w1 = Wo + (size_t)(tg * 64 + 1 * 16 + (lane & 15)) * D + oB;
    const f16* w2 = Wo + (size_t)(tg * 64 + 2 * 16 + (lane & 15)) * D + oB;
    const f16* w3 = Wo + (size_t)(tg * 64 + 3 * 16 + (lane & 15)) * D + oB;

    v8f acc0 = {}, acc1 = {}, acc2 = {}, acc3 = {};
    for (int kk0 = 0; kk0 < D; kk0 += 32) {
        const v16h a = ldA16(arow + kk0, lane);
        acc0 = WMMA16(a, *(const v16h*)(w0 + kk0), acc0);
        acc1 = WMMA16(a, *(const v16h*)(w1 + kk0), acc1);
        acc2 = WMMA16(a, *(const v16h*)(w2 + kk0), acc2);
        acc3 = WMMA16(a, *(const v16h*)(w3 + kk0), acc3);
    }

    v8f accs[4] = {acc0, acc1, acc2, acc3};
#pragma unroll
    for (int t = 0; t < 4; ++t) {
        const int n  = tg * 64 + t * 16 + (lane & 15);
        const float bias = bo[n];
#pragma unroll
        for (int j = 0; j < 8; ++j) {
            const int M = (lane & 16) ? j + 8 : j;
            out[(size_t)(mtile * 16 + M) * D + n] = accs[t][j] + bias;
        }
    }
}

// ---- host launch ------------------------------------------------------------
extern "C" void kernel_launch(void* const* d_in, const int* in_sizes, int n_in,
                              void* d_out, int out_size, void* d_ws, size_t ws_size,
                              hipStream_t stream) {
    const float* Q  = (const float*)d_in[0];
    const float* K  = (const float*)d_in[1];
    const float* V  = (const float*)d_in[2];
    const int*   pm = (const int*)d_in[3];
    // d_in[4] = look_ahead_mask (causal) -- computed analytically in-kernel
    const float* Wq = (const float*)d_in[5];
    const float* bq = (const float*)d_in[6];
    const float* Wk = (const float*)d_in[7];
    const float* bk = (const float*)d_in[8];
    const float* Wv = (const float*)d_in[9];
    const float* bv = (const float*)d_in[10];
    const float* Wo = (const float*)d_in[11];
    const float* bo = (const float*)d_in[12];

    const size_t xel = (size_t)Bsz * S * D;       // 4,194,304 (activations)
    const size_t wel = (size_t)D * D;             // 1,048,576 (weights)

    f16* xq16  = (f16*)d_ws;                      // converted inputs
    f16* xk16  = xq16 + xel;
    f16* xv16  = xk16 + xel;
    f16* wq16  = xv16 + xel;                      // converted weights
    f16* wk16  = wq16 + wel;
    f16* wv16  = wk16 + wel;
    f16* wo16  = wv16 + wel;
    f16* q_ws  = wo16 + wel;                      // projected heads
    f16* k_ws  = q_ws + xel;
    f16* vT_ws = k_ws + xel;
    f16* att_ws = vT_ws + xel;                    // attention output

    // one-time f32 -> f16 conversion pass (bandwidth-bound, ~100 MB total)
    const int x4 = (int)(xel / 4), w4 = (int)(wel / 4);
    cvt_f32_f16_kernel<<<x4 / 256, 256, 0, stream>>>(Q,  xq16, x4);
    cvt_f32_f16_kernel<<<x4 / 256, 256, 0, stream>>>(K,  xk16, x4);
    cvt_f32_f16_kernel<<<x4 / 256, 256, 0, stream>>>(V,  xv16, x4);
    cvt_f32_f16_kernel<<<w4 / 256, 256, 0, stream>>>(Wq, wq16, w4);
    cvt_f32_f16_kernel<<<w4 / 256, 256, 0, stream>>>(Wk, wk16, w4);
    cvt_f32_f16_kernel<<<w4 / 256, 256, 0, stream>>>(Wv, wv16, w4);
    cvt_f32_f16_kernel<<<w4 / 256, 256, 0, stream>>>(Wo, wo16, w4);

    qkv_proj_kernel<<<1536, 256, 0, stream>>>(xq16, xk16, xv16,
                                              wq16, bq, wk16, bk, wv16, bv,
                                              q_ws, k_ws, vT_ws);
    attn_kernel<<<512, 256, 0, stream>>>(q_ws, k_ws, vT_ws, pm, att_ws);
    out_proj_kernel<<<512, 256, 0, stream>>>(att_ws, wo16, bo, (float*)d_out);
}
